// TransformerXLLayer_13786845020973
// MI455X (gfx1250) — matrix operand order repaired and checked by reading, not compiled, not run on hardware
//
#include <hip/hip_runtime.h>

// ---------------------------------------------------------------------------
// Types for CDNA5 WMMA (wave32, 16x16x32 bf16 -> f32 accum)
// ---------------------------------------------------------------------------
typedef __bf16 bf16_t;
typedef __attribute__((ext_vector_type(16))) __bf16 v16bf;
typedef __attribute__((ext_vector_type(8)))  float  v8f;

union FragU { uint4 u4[2]; v16bf v; };
union Pack8 { uint4 u; bf16_t h[8]; };

enum {
  EPI_F32 = 0,        // f32 C, batched via sC
  EPI_BF16,           // bf16 C, batched via sC
  EPI_Q,              // C=q (bf16), C2=q+bias (bf16)
  EPI_KT,             // scatter to kT[(b*16+h)*64+d][1024 j]
  EPI_V,              // scatter to v[(b*16+h)*1024+j][64 d]
  EPI_AV,             // scatter heads back: out[(i*8+b)*1024 + h*64+d]
  EPI_BIAS_RES_F32,   // f32 C = acc + bias[col] + residual[row,col]
  EPI_RELU_BF16       // bf16 C = relu(acc + bias[col])
};

struct GemmArgs {
  const bf16_t* A; long lda; long sA;   // row-major A, per-batch elem stride
  const bf16_t* B; long ldb; long sB;   // row-major B (K x N)
  void* C; void* C2;
  long ldc; long sC;
  const float* bias;
  const float* residual;
  int mode; int Kdim;
};

// ---------------------------------------------------------------------------
// Generic bf16 WMMA GEMM.
//  Block tile BM x BN, 8 waves, each wave owns WM x WN (MF x NF WMMA frags).
//  Double-buffered LDS; A tile staged with global_load_async_to_lds_b128
//  (ASYNCcnt), B tile staged load+transpose so both operands come back as
//  contiguous 16B ds_load_b128 matching the ISA 16-bit A(16x32)/B(32x16)
//  VGPR layouts. One barrier per k-step; async copy overlaps the WMMAs.
// ---------------------------------------------------------------------------
template <int BM, int BN, int WM, int WN>
__global__ __launch_bounds__(256) void gemm_bf16_wmma(GemmArgs g) {
  constexpr int MF = WM / 16, NF = WN / 16;
  constexpr int WCOL = BN / WN;
  __shared__ bf16_t As[2][BM * 32];   // [m][k]
  __shared__ bf16_t Bs[2][BN * 32];   // transposed: [n][k]

  const int tid  = threadIdx.x;
  const int lane = tid & 31;
  const int wave = tid >> 5;
  const int wm   = wave / WCOL;
  const int wn   = wave % WCOL;
  const int lm   = lane & 15;
  const int kh   = lane >> 4;          // half-wave selector per ISA layout
  const int z    = blockIdx.z;

  const bf16_t* gA = g.A + (long)z * g.sA;
  const bf16_t* gB = g.B + (long)z * g.sB;
  const long rowBlk = (long)blockIdx.y * BM;
  const long colBlk = (long)blockIdx.x * BN;
  const int nk = g.Kdim >> 5;

  v8f acc[MF][NF];
#pragma unroll
  for (int mf = 0; mf < MF; ++mf)
#pragma unroll
    for (int nf = 0; nf < NF; ++nf) acc[mf][nf] = (v8f){0, 0, 0, 0, 0, 0, 0, 0};

  // LDS byte offset of As (flat shared address truncates to LDS offset)
  const unsigned asBase = (unsigned)(size_t)(&As[0][0]);

  auto stage = [&](int kt, int buf) {
    // A tile: BM x 32, async global->LDS 16B per lane (ASYNCcnt tracked)
#pragma unroll
    for (int e = tid * 8; e < BM * 32; e += 2048) {
      const int m = e >> 5, k2 = e & 31;
      const bf16_t* src = gA + (rowBlk + m) * g.lda + (long)kt * 32 + k2;
      const unsigned dst = asBase + (unsigned)(buf * BM * 32 + e) * 2u;
      asm volatile("global_load_async_to_lds_b128 %0, %1, off"
                   :: "v"(dst), "v"(src) : "memory");
      __builtin_prefetch(src + 32, 0, 1);  // next k-tile (speculative ok)
    }
    // B tile: 32 x BN, load 16B then transpose-scatter into [n][k]
#pragma unroll
    for (int e = tid * 8; e < BN * 32; e += 2048) {
      const int kb = e / BN, n0 = e % BN;
      const bf16_t* src = gB + ((long)kt * 32 + kb) * g.ldb + colBlk + n0;
      Pack8 p;
      p.u = *(const uint4*)src;
#pragma unroll
      for (int xx = 0; xx < 8; ++xx) Bs[buf][(n0 + xx) * 32 + kb] = p.h[xx];
      __builtin_prefetch(src + 32 * g.ldb, 0, 1);
    }
  };

  auto compute = [&](int buf) {
    FragU a[MF], b[NF];
#pragma unroll
    for (int mf = 0; mf < MF; ++mf) {
      const bf16_t* arow = &As[buf][(wm * WM + mf * 16 + lm) * 32];
      a[mf].u4[0] = *(const uint4*)(arow + kh * 8);
      a[mf].u4[1] = *(const uint4*)(arow + 16 + kh * 8);
    }
#pragma unroll
    for (int nf = 0; nf < NF; ++nf) {
      const bf16_t* brow = &Bs[buf][(wn * WN + nf * 16 + lm) * 32];
      b[nf].u4[0] = *(const uint4*)(brow + kh * 8);
      b[nf].u4[1] = *(const uint4*)(brow + 16 + kh * 8);
    }
#pragma unroll
    for (int mf = 0; mf < MF; ++mf)
#pragma unroll
      for (int nf = 0; nf < NF; ++nf)
        acc[mf][nf] = __builtin_amdgcn_wmma_f32_16x16x32_bf16(
            false, a[mf].v, false, b[nf].v, (short)0, acc[mf][nf], false, false);
  };

  stage(0, 0);
  asm volatile("s_wait_asynccnt 0" ::: "memory");
  __syncthreads();
  for (int kt = 0; kt < nk; ++kt) {
    const int cur = kt & 1;
    if (kt + 1 < nk) stage(kt + 1, cur ^ 1);  // overlaps with WMMAs below
    compute(cur);
    asm volatile("s_wait_asynccnt 0" ::: "memory");
    __syncthreads();
  }

  // C layout: VGPR r -> row r (lanes 0-15) / r+8 (lanes 16-31), col = lane&15
#pragma unroll
  for (int mf = 0; mf < MF; ++mf) {
    const long rowBase = rowBlk + wm * WM + mf * 16 + (kh << 3);
#pragma unroll
    for (int nf = 0; nf < NF; ++nf) {
      const long colg = colBlk + wn * WN + nf * 16 + lm;
      v8f a8 = acc[mf][nf];
#pragma unroll
      for (int r = 0; r < 8; ++r) {
        const long rowg = rowBase + r;
        const float val = a8[r];
        switch (g.mode) {
          case EPI_F32:
            ((float*)g.C)[(long)z * g.sC + rowg * g.ldc + colg] = val;
            break;
          case EPI_BF16:
            ((bf16_t*)g.C)[(long)z * g.sC + rowg * g.ldc + colg] = (bf16_t)val;
            break;
          case EPI_Q: {
            const long idx = rowg * 1024 + colg;
            ((bf16_t*)g.C)[idx]  = (bf16_t)val;
            ((bf16_t*)g.C2)[idx] = (bf16_t)(val + g.bias[colg]);
          } break;
          case EPI_KT: {
            const long b = rowg & 7, j = rowg >> 3, h = colg >> 6, d = colg & 63;
            ((bf16_t*)g.C)[(((b * 16 + h) * 64 + d) << 10) + j] = (bf16_t)val;
          } break;
          case EPI_V: {
            const long b = rowg & 7, j = rowg >> 3, h = colg >> 6, d = colg & 63;
            ((bf16_t*)g.C)[((((b * 16 + h) << 10) + j) << 6) + d] = (bf16_t)val;
          } break;
          case EPI_AV: {
            const long b = z >> 4, h = z & 15;
            ((bf16_t*)g.C)[((rowg * 8 + b) << 10) + h * 64 + colg] = (bf16_t)val;
          } break;
          case EPI_BIAS_RES_F32: {
            const long idx = rowg * g.ldc + colg;
            ((float*)g.C)[idx] = val + g.bias[colg] + g.residual[idx];
          } break;
          case EPI_RELU_BF16: {
            const float u = val + g.bias[colg];
            ((bf16_t*)g.C)[rowg * g.ldc + colg] = (bf16_t)(u > 0.f ? u : 0.f);
          } break;
        }
      }
    }
  }
}

// ---------------------------------------------------------------------------
// LayerNorm over D=1024 (f32 in -> bf16 out), one block per row
// ---------------------------------------------------------------------------
__global__ __launch_bounds__(256) void layernorm_bf16(const float* __restrict__ in,
                                                      const float* __restrict__ gg,
                                                      const float* __restrict__ bb,
                                                      bf16_t* __restrict__ out) {
  const int D = 1024;
  const long row = blockIdx.x;
  const float* x = in + row * D;
  const int tid = threadIdx.x;
  float v[4], s = 0.f, s2 = 0.f;
#pragma unroll
  for (int u = 0; u < 4; ++u) {
    v[u] = x[tid + u * 256];
    s += v[u];
    s2 += v[u] * v[u];
  }
  __shared__ float rs[256], rq[256];
  rs[tid] = s; rq[tid] = s2;
  __syncthreads();
  for (int st = 128; st > 0; st >>= 1) {
    if (tid < st) { rs[tid] += rs[tid + st]; rq[tid] += rq[tid + st]; }
    __syncthreads();
  }
  const float mu  = rs[0] * (1.f / 1024.f);
  const float var = rq[0] * (1.f / 1024.f) - mu * mu;
  const float inv = rsqrtf(var + 1e-5f);
#pragma unroll
  for (int u = 0; u < 4; ++u) {
    const int c = tid + u * 256;
    out[row * D + c] = (bf16_t)((v[u] - mu) * inv * gg[c] + bb[c]);
  }
}

// ---------------------------------------------------------------------------
// Elementwise f32 -> bf16
// ---------------------------------------------------------------------------
__global__ void f32_to_bf16_k(const float* __restrict__ in, bf16_t* __restrict__ out,
                              long n) {
  long i = (long)blockIdx.x * blockDim.x + threadIdx.x;
  const long stride = (long)gridDim.x * blockDim.x;
  for (; i < n; i += stride) out[i] = (bf16_t)in[i];
}

// Build per-head transposed bf16 kpe slice: out[(h*64+d)*1536 + t]
// from key_pos_emb[(3072+t)*1024 + h*64+d]  (slice P-K .. P+S-1, len 1536)
__global__ void build_kpeT(const float* __restrict__ kpe, bf16_t* __restrict__ out) {
  const long total = 16L * 64 * 1536;
  long idx = (long)blockIdx.x * blockDim.x + threadIdx.x;
  if (idx >= total) return;
  const long t = idx % 1536, hd = idx / 1536;
  out[idx] = (bf16_t)kpe[(3072 + t) * 1024 + hd];
}

// ---------------------------------------------------------------------------
// Softmax with TXL rel-shift gather:
//   score[j] = (ac[j] + Bpos[i*8+b, j+512-i] + kpb[j+512-i, h]) / 8,
//   masked where j > i+512. Writes bf16 probabilities.
// grid = (512 i, 128 z=b*16+h), block = 256
// ---------------------------------------------------------------------------
__global__ __launch_bounds__(256) void softmax_rel(const float* __restrict__ scores,
                                                   const bf16_t* __restrict__ bpos,
                                                   const float* __restrict__ kpb_full,
                                                   bf16_t* __restrict__ attn) {
  const int i = blockIdx.x;
  const int z = blockIdx.y;
  const int b = z >> 4, h = z & 15;
  const int tid = threadIdx.x;
  const float* srow = scores + ((long)z * 512 + i) * 1024;
  const bf16_t* brow = bpos + ((long)h * 4096 + (long)i * 8 + b) * 1536;
  bf16_t* arow = attn + ((long)z * 512 + i) * 1024;
  const int jmax = i + 512;

  float vals[4], lmax = -1e30f;
#pragma unroll
  for (int u = 0; u < 4; ++u) {
    const int j = tid + u * 256;
    float v = -1e30f;
    if (j <= jmax) {
      const int t = j + 512 - i;  // in [1,1535], never the shift pad
      v = (srow[j] + (float)brow[t] + kpb_full[(long)(3072 + t) * 16 + h]) * 0.125f;
    }
    vals[u] = v;
    lmax = fmaxf(lmax, v);
  }
  __shared__ float red[256];
  red[tid] = lmax; __syncthreads();
  for (int st = 128; st > 0; st >>= 1) {
    if (tid < st) red[tid] = fmaxf(red[tid], red[tid + st]);
    __syncthreads();
  }
  const float rmax = red[0];
  __syncthreads();
  float e[4], lsum = 0.f;
#pragma unroll
  for (int u = 0; u < 4; ++u) { e[u] = __expf(vals[u] - rmax); lsum += e[u]; }
  red[tid] = lsum; __syncthreads();
  for (int st = 128; st > 0; st >>= 1) {
    if (tid < st) red[tid] += red[tid + st];
    __syncthreads();
  }
  const float inv = 1.f / red[0];
#pragma unroll
  for (int u = 0; u < 4; ++u) arow[tid + u * 256] = (bf16_t)(e[u] * inv);
}

// ---------------------------------------------------------------------------
// Host side
// ---------------------------------------------------------------------------
static inline void launch_gemm(hipStream_t s, const bf16_t* A, long lda, long sA,
                               const bf16_t* Bm, long ldb, long sB, void* C, void* C2,
                               long ldc, long sC, const float* bias,
                               const float* residual, int mode, int M, int N,
                               int Kdim, int batches) {
  GemmArgs g;
  g.A = A; g.lda = lda; g.sA = sA;
  g.B = Bm; g.ldb = ldb; g.sB = sB;
  g.C = C; g.C2 = C2; g.ldc = ldc; g.sC = sC;
  g.bias = bias; g.residual = residual;
  g.mode = mode; g.Kdim = Kdim;
  if (N % 128 == 0) {
    dim3 grid(N / 128, M / 128, batches);
    gemm_bf16_wmma<128, 128, 32, 64><<<grid, dim3(256), 0, s>>>(g);
  } else {
    dim3 grid(N / 64, M / 128, batches);
    gemm_bf16_wmma<128, 64, 32, 32><<<grid, dim3(256), 0, s>>>(g);
  }
}

extern "C" void kernel_launch(void* const* d_in, const int* in_sizes, int n_in,
                              void* d_out, int out_size, void* d_ws, size_t ws_size,
                              hipStream_t stream) {
  (void)in_sizes; (void)n_in; (void)out_size; (void)ws_size;
  const float* x      = (const float*)d_in[0];
  const float* mem    = (const float*)d_in[1];
  // d_in[2] = mask (unused; mask is analytic: j <= i + MEM)
  const float* w_q    = (const float*)d_in[3];
  const float* w_k    = (const float*)d_in[4];
  const float* w_v    = (const float*)d_in[5];
  const float* w_o    = (const float*)d_in[6];
  const float* b_o    = (const float*)d_in[7];
  const float* kpe    = (const float*)d_in[8];
  const float* kpb    = (const float*)d_in[9];
  const float* qpb    = (const float*)d_in[10];
  const float* ln1_g  = (const float*)d_in[11];
  const float* ln1_b  = (const float*)d_in[12];
  const float* ln2_g  = (const float*)d_in[13];
  const float* ln2_b  = (const float*)d_in[14];
  const float* ff_w1  = (const float*)d_in[15];
  const float* ff_b1  = (const float*)d_in[16];
  const float* ff_w2  = (const float*)d_in[17];
  const float* ff_b2  = (const float*)d_in[18];
  float* out          = (float*)d_out;

  // bump allocator over d_ws
  char* wp = (char*)d_ws;
  auto alloc = [&](size_t bytes) -> void* {
    void* p = wp;
    wp += (bytes + 255) & ~(size_t)255;
    return p;
  };
  bf16_t* wq_b  = (bf16_t*)alloc(1024L * 1024 * 2);
  bf16_t* wk_b  = (bf16_t*)alloc(1024L * 1024 * 2);
  bf16_t* wv_b  = (bf16_t*)alloc(1024L * 1024 * 2);
  bf16_t* wo_b  = (bf16_t*)alloc(1024L * 1024 * 2);
  bf16_t* fw1_b = (bf16_t*)alloc(1024L * 4096 * 2);
  bf16_t* fw2_b = (bf16_t*)alloc(4096L * 1024 * 2);
  bf16_t* kpeT  = (bf16_t*)alloc(16L * 64 * 1536 * 2);
  bf16_t* mz    = (bf16_t*)alloc(8192L * 1024 * 2);   // [j*8+b][D], mem_n ++ x
  bf16_t* qB    = (bf16_t*)alloc(4096L * 1024 * 2);   // q
  bf16_t* qbB   = (bf16_t*)alloc(4096L * 1024 * 2);   // q + query_pos_bias
  bf16_t* kT    = (bf16_t*)alloc(8192L * 1024 * 2);   // [(b*16+h)*64+d][j]
  bf16_t* vB    = (bf16_t*)alloc(8192L * 1024 * 2);   // [(b*16+h)*1024+j][d]
  float*  sc    = (float*)alloc(128L * 512 * 1024 * 4); // ac scores per (b,h)
  bf16_t* bpos  = (bf16_t*)alloc(16L * 4096 * 1536 * 2);// q . kpe^T per h
  bf16_t* attn  = (bf16_t*)alloc(128L * 512 * 1024 * 2);// probs
  bf16_t* aout  = (bf16_t*)alloc(4096L * 1024 * 2);   // heads merged
  float*  x2    = (float*)alloc(4096L * 1024 * 4);    // x + attn proj
  bf16_t* z2    = (bf16_t*)alloc(4096L * 1024 * 2);   // LN2(x2)
  bf16_t* ffh   = (bf16_t*)alloc(4096L * 4096 * 2);   // relu hidden

  // 1) weight conversions + positional table
  f32_to_bf16_k<<<2048, 256, 0, stream>>>(w_q, wq_b, 1024L * 1024);
  f32_to_bf16_k<<<2048, 256, 0, stream>>>(w_k, wk_b, 1024L * 1024);
  f32_to_bf16_k<<<2048, 256, 0, stream>>>(w_v, wv_b, 1024L * 1024);
  f32_to_bf16_k<<<2048, 256, 0, stream>>>(w_o, wo_b, 1024L * 1024);
  f32_to_bf16_k<<<2048, 256, 0, stream>>>(ff_w1, fw1_b, 1024L * 4096);
  f32_to_bf16_k<<<2048, 256, 0, stream>>>(ff_w2, fw2_b, 4096L * 1024);
  build_kpeT<<<(16 * 64 * 1536 + 255) / 256, 256, 0, stream>>>(kpe, kpeT);

  // 2) m_z = [LN(mem); x] in bf16
  layernorm_bf16<<<4096, 256, 0, stream>>>(mem, ln1_g, ln1_b, mz);
  f32_to_bf16_k<<<2048, 256, 0, stream>>>(x, mz + 4096L * 1024, 4096L * 1024);

  // 3) projections
  launch_gemm(stream, mz + 4096L * 1024, 1024, 0, wq_b, 1024, 0, qB, qbB, 1024, 0,
              qpb, nullptr, EPI_Q, 4096, 1024, 1024, 1);
  launch_gemm(stream, mz, 1024, 0, wk_b, 1024, 0, kT, nullptr, 0, 0, nullptr,
              nullptr, EPI_KT, 8192, 1024, 1024, 1);
  launch_gemm(stream, mz, 1024, 0, wv_b, 1024, 0, vB, nullptr, 0, 0, nullptr,
              nullptr, EPI_V, 8192, 1024, 1024, 1);

  // 4) content scores ac: per (b,h): (q+qpb)[512x64] x kT[64x1024]
  launch_gemm(stream, qbB, 8192, 64, kT, 1024, 64L * 1024, sc, nullptr, 1024,
              512L * 1024, nullptr, nullptr, EPI_F32, 512, 1024, 64, 128);

  // 5) positional scores: per h: q[4096x64] x kpeT[64x1536]
  launch_gemm(stream, qB, 1024, 64, kpeT, 1536, 64L * 1536, bpos, nullptr, 1536,
              4096L * 1536, nullptr, nullptr, EPI_BF16, 4096, 1536, 64, 16);

  // 6) rel-shift + mask + softmax
  softmax_rel<<<dim3(512, 128), 256, 0, stream>>>(sc, bpos, kpb, attn);

  // 7) attn x V: per (b,h): [512x1024] x [1024x64] -> merged heads
  launch_gemm(stream, attn, 1024, 512L * 1024, vB, 64, 1024L * 64, aout, nullptr,
              0, 0, nullptr, nullptr, EPI_AV, 512, 64, 1024, 128);

  // 8) output projection + residual: x2 = x + aout @ w_o + b_o
  launch_gemm(stream, aout, 1024, 0, wo_b, 1024, 0, x2, nullptr, 1024, 0, b_o, x,
              EPI_BIAS_RES_F32, 4096, 1024, 1024, 1);

  // 9) FFN
  layernorm_bf16<<<4096, 256, 0, stream>>>(x2, ln2_g, ln2_b, z2);
  launch_gemm(stream, z2, 1024, 0, fw1_b, 4096, 0, ffh, nullptr, 4096, 0, ff_b1,
              nullptr, EPI_RELU_BF16, 4096, 4096, 1024, 1);
  launch_gemm(stream, ffh, 4096, 0, fw2_b, 1024, 0, out, nullptr, 1024, 0, ff_b2,
              x2, EPI_BIAS_RES_F32, 4096, 1024, 4096, 1);
}
